// AttentionLayer_77077483094558
// MI455X (gfx1250) — matrix operand order repaired and checked by reading, not compiled, not run on hardware
//
#include <hip/hip_runtime.h>
#include <hip/hip_bf16.h>

// ---------------------------------------------------------------------------
// Hypergraph attention: out[e, k] = softmax_k( x[idx[e], :] . W[k, :] )
// E = 500000 edges, D = 1024 features, 32 attention heads.
//
// MI455X strategy:
//  * Pre-convert x (400MB f32 -> 200MB bf16) into workspace: the gather
//    working set then fits (almost) in the 192MB L2, cutting HBM gather
//    traffic from ~2GB to ~200-400MB.
//  * 16-edge tiles per wave32; logits via v_wmma_f32_16x16x32_bf16
//    (two 16-wide N tiles for the 32 heads), f32 accumulate.
//  * Softmax epilogue through a per-wave LDS transpose: one shfl_xor(16)
//    per reduction instead of 64 ds_bpermutes; coalesced b128 NT stores.
// ---------------------------------------------------------------------------

typedef __attribute__((ext_vector_type(16))) __bf16 v16bf;
typedef __attribute__((ext_vector_type(8)))  float  v8f;
typedef __attribute__((ext_vector_type(4)))  float  v4f;

#define IN_F 1024
#define LROW 36   // LDS row stride in dwords: 16B-aligned, bank-conflict-light

union Frag16 {
    v16bf  v;
    uint4  q[2];
    __bf16 e[16];
};

__device__ __forceinline__ __bf16 tobf(float f) { return (__bf16)f; }

__device__ __forceinline__ float vmax4(v4f v) {
    return fmaxf(fmaxf(v[0], v[1]), fmaxf(v[2], v[3]));
}
__device__ __forceinline__ float vsum4(v4f v) {
    return (v[0] + v[1]) + (v[2] + v[3]);
}
__device__ __forceinline__ v4f vexp4(v4f v, float m) {
    v4f r;
    r[0] = __expf(v[0] - m); r[1] = __expf(v[1] - m);
    r[2] = __expf(v[2] - m); r[3] = __expf(v[3] - m);
    return r;
}

// f32 -> bf16 packed conversion (x and W). NT loads: source is single-use
// streaming data; destination stays regular-temporal so it lives in L2.
__global__ __launch_bounds__(256) void cvt_f32_bf16_kernel(
    const float* __restrict__ in, __bf16* __restrict__ out, long long n)
{
    long long i = ((long long)blockIdx.x * blockDim.x + threadIdx.x) * 8;
    if (i >= n) return;
    if (i + 8 > n) {
        for (long long k = i; k < n; ++k) out[k] = tobf(in[k]);
        return;
    }
    v4f a = __builtin_nontemporal_load((const v4f*)(in + i));
    v4f b = __builtin_nontemporal_load((const v4f*)(in + i + 4));
    union { __bf16 e[8]; uint4 q; } u;
    u.e[0] = tobf(a[0]); u.e[1] = tobf(a[1]); u.e[2] = tobf(a[2]); u.e[3] = tobf(a[3]);
    u.e[4] = tobf(b[0]); u.e[5] = tobf(b[1]); u.e[6] = tobf(b[2]); u.e[7] = tobf(b[3]);
    *(uint4*)(out + i) = u.q;
}

// Main attention kernel. One wave32 per 16-edge tile, 8 waves per block.
// PRE=true : A from bf16 x-copy in ws, B from bf16 W-copy in ws.
// PRE=false: gather f32 x / f32 W and convert in-registers.
template<bool PRE>
__global__ __launch_bounds__(256) void attn_wmma_kernel(
    const float*   __restrict__ x,
    const __bf16*  __restrict__ xbf,
    const int*     __restrict__ eidx,
    const float*   __restrict__ W,
    const __bf16*  __restrict__ wbf,
    float*         __restrict__ out,
    int E, int ntiles)
{
    const int wave = threadIdx.x >> 5;
    const int lane = threadIdx.x & 31;
    const int tile = blockIdx.x * (blockDim.x >> 5) + wave;
    if (tile >= ntiles) return;          // uniform per wave: EXEC stays all-1s

    const int m = lane & 15;             // edge row within tile (A) / att col (B,D)
    const int h = lane >> 4;             // half-wave selector

    int ei = tile * 16 + m;
    if (ei >= E) ei = E - 1;             // clamp padded lanes (stores guarded later)
    const long long node = (long long)eidx[ei];

    const float*  xrow  = x   + node * (long long)IN_F;
    const __bf16* xbrow = xbf + node * (long long)IN_F;
    const float*  w0    = W   + (long long)m * IN_F;          // att rows 0..15
    const float*  w1    = W   + (long long)(m + 16) * IN_F;   // att rows 16..31
    const __bf16* wb0   = wbf + (long long)m * IN_F;
    const __bf16* wb1   = wbf + (long long)(m + 16) * IN_F;

    // Operand k order used here (wave32, contiguous; not AMD's table order):
    //  A 16x32 bf16: lane m holds K = [16h .. 16h+15] contiguous in e[0..15]
    //  B 32x16 bf16: lane n holds K = [16h .. 16h+15] contiguous in e[0..15]
    const int aoff = 16 * h;
    const int boff = 16 * h;

    v8f acc0 = {};   // logits for att heads 0..15
    v8f acc1 = {};   // logits for att heads 16..31

    #pragma unroll 4
    for (int d0 = 0; d0 < IN_F; d0 += 32) {
        Frag16 a, b0, b1;
        if (PRE) {
            a.q[0]  = *(const uint4*)(xbrow + d0 + aoff);
            a.q[1]  = *(const uint4*)(xbrow + d0 + aoff + 8);
            b0.q[0] = *(const uint4*)(wb0 + d0 + boff);
            b0.q[1] = *(const uint4*)(wb0 + d0 + boff + 8);
            b1.q[0] = *(const uint4*)(wb1 + d0 + boff);
            b1.q[1] = *(const uint4*)(wb1 + d0 + boff + 8);
        } else {
            float4 f0 = *(const float4*)(xrow + d0 + aoff);
            float4 f1 = *(const float4*)(xrow + d0 + aoff + 4);
            float4 f2 = *(const float4*)(xrow + d0 + aoff + 8);
            float4 f3 = *(const float4*)(xrow + d0 + aoff + 12);
            a.e[0]=tobf(f0.x); a.e[1]=tobf(f0.y); a.e[2]=tobf(f0.z); a.e[3]=tobf(f0.w);
            a.e[4]=tobf(f1.x); a.e[5]=tobf(f1.y); a.e[6]=tobf(f1.z); a.e[7]=tobf(f1.w);
            a.e[8]=tobf(f2.x); a.e[9]=tobf(f2.y); a.e[10]=tobf(f2.z); a.e[11]=tobf(f2.w);
            a.e[12]=tobf(f3.x); a.e[13]=tobf(f3.y); a.e[14]=tobf(f3.z); a.e[15]=tobf(f3.w);
            #pragma unroll
            for (int t = 0; t < 4; ++t) {
                float4 g0 = *(const float4*)(w0 + d0 + boff + 4 * t);
                float4 g1 = *(const float4*)(w1 + d0 + boff + 4 * t);
                b0.e[4*t+0]=tobf(g0.x); b0.e[4*t+1]=tobf(g0.y);
                b0.e[4*t+2]=tobf(g0.z); b0.e[4*t+3]=tobf(g0.w);
                b1.e[4*t+0]=tobf(g1.x); b1.e[4*t+1]=tobf(g1.y);
                b1.e[4*t+2]=tobf(g1.z); b1.e[4*t+3]=tobf(g1.w);
            }
        }
        acc0 = __builtin_amdgcn_wmma_f32_16x16x32_bf16(
                   false, a.v, false, b0.v, (short)0, acc0, false, false);
        acc1 = __builtin_amdgcn_wmma_f32_16x16x32_bf16(
                   false, a.v, false, b1.v, (short)0, acc1, false, false);
    }

    // ---- Softmax epilogue via per-wave LDS tile -------------------------
    // D layout: lane holds column n = m, rows r = j + 8h for VGPR j.
    // Scatter logits to this wave's private LDS tile, then re-assign lanes:
    // lane -> (edge r2 = m, column half cb = 16h) so each lane owns 16
    // contiguous floats. Only one shfl_xor(16) per reduction is needed to
    // merge the two halves of each edge.
    __shared__ __align__(16) float lds[8][16 * LROW];
    float* L = lds[wave];

    #pragma unroll
    for (int j = 0; j < 8; ++j) {
        const int r = j + 8 * h;
        L[r * LROW + m]      = acc0[j];
        L[r * LROW + 16 + m] = acc1[j];
    }
    // Same-wave LDS ops are tracked by DScnt; make the scatter visible
    // before the cross-lane re-read.
    asm volatile("s_wait_dscnt 0x0" ::: "memory");

    const int r2 = m;                    // edge this lane reduces
    const int cb = 16 * h;               // column half this lane owns
    const float* Lr = &L[r2 * LROW + cb];
    v4f c0 = *(const v4f*)(Lr);
    v4f c1 = *(const v4f*)(Lr + 4);
    v4f c2 = *(const v4f*)(Lr + 8);
    v4f c3 = *(const v4f*)(Lr + 12);

    float mx = fmaxf(fmaxf(vmax4(c0), vmax4(c1)), fmaxf(vmax4(c2), vmax4(c3)));
    mx = fmaxf(mx, __shfl_xor(mx, 16, 32));

    c0 = vexp4(c0, mx); c1 = vexp4(c1, mx);
    c2 = vexp4(c2, mx); c3 = vexp4(c3, mx);

    float s = (vsum4(c0) + vsum4(c1)) + (vsum4(c2) + vsum4(c3));
    s += __shfl_xor(s, 16, 32);
    const float inv = 1.0f / s;
    c0 *= inv; c1 *= inv; c2 *= inv; c3 *= inv;

    if (tile * 16 + r2 < E) {
        float* op = out + (long long)(tile * 16 + r2) * 32 + cb;
        // Output is write-once: NT stores keep the bf16 x copy resident in L2.
        __builtin_nontemporal_store(c0, (v4f*)(op));
        __builtin_nontemporal_store(c1, (v4f*)(op + 4));
        __builtin_nontemporal_store(c2, (v4f*)(op + 8));
        __builtin_nontemporal_store(c3, (v4f*)(op + 12));
    }
}

extern "C" void kernel_launch(void* const* d_in, const int* in_sizes, int n_in,
                              void* d_out, int out_size, void* d_ws, size_t ws_size,
                              hipStream_t stream) {
    const float* x   = (const float*)d_in[0];
    const int*   idx = (const int*)d_in[1];
    const float* W   = (const float*)d_in[2];
    float*       out = (float*)d_out;

    const int E = in_sizes[1];                       // 500000 edges
    const int ntiles = (E + 15) / 16;

    const long long xn = (long long)in_sizes[0];     // N_NODES * 1024
    const long long wn = (long long)in_sizes[2];     // 32 * 1024
    const size_t x_bytes = (size_t)xn * sizeof(__bf16);
    const size_t w_bytes = (size_t)wn * sizeof(__bf16);

    const bool pre = (d_ws != nullptr) && (ws_size >= x_bytes + w_bytes);

    __bf16* xbf = (__bf16*)d_ws;
    __bf16* wbf = (__bf16*)((char*)d_ws + x_bytes);  // x_bytes is 16B-aligned

    const int cvt_blk = 256;
    const int attn_blk = 256;                        // 8 waves -> 8 tiles per block
    const int attn_grid = (ntiles + 7) / 8;

    if (pre) {
        unsigned xg = (unsigned)((xn / 8 + cvt_blk - 1) / cvt_blk);
        unsigned wg = (unsigned)((wn / 8 + cvt_blk - 1) / cvt_blk);
        cvt_f32_bf16_kernel<<<xg, cvt_blk, 0, stream>>>(x, xbf, xn);
        cvt_f32_bf16_kernel<<<wg, cvt_blk, 0, stream>>>(W, wbf, wn);
        attn_wmma_kernel<true><<<attn_grid, attn_blk, 0, stream>>>(
            x, xbf, idx, W, wbf, out, E, ntiles);
    } else {
        attn_wmma_kernel<false><<<attn_grid, attn_blk, 0, stream>>>(
            x, xbf, idx, W, wbf, out, E, ntiles);
    }
}
